// WindowAttention_82609400971400
// MI455X (gfx1250) — matrix operand order repaired and checked by reading, not compiled, not run on hardware
//
#include <hip/hip_runtime.h>
#include <hip/hip_bf16.h>

typedef __attribute__((ext_vector_type(16))) _Float16 v16h;
typedef __attribute__((ext_vector_type(8)))  _Float16 v8h;
typedef __attribute__((ext_vector_type(8)))  float    v8f;

#define SEQ      49
#define SEQP     64
#define DIM      512
#define NHEAD    16
#define HEADD    32
#define SXS      520                       // f16 stride for x / attn-out LDS tiles (bank-stagger)
#define SX_BYTES (SEQP * SXS * 2)          // 66560
#define QKP_BYTES 9216                     // union{ q[64][32], k[64][32] } / P[64][72] f16
#define VT_BYTES  4096                     // vT[32][64] f16
#define WAVE_BYTES (QKP_BYTES + VT_BYTES)  // 13312
#define SMEM_TOTAL (2 * SX_BYTES + 8 * WAVE_BYTES)  // 239616
#define SCALE 0.17677669529663687f         // 1/sqrt(32)

// ---- WMMA fragment helpers (CDNA5 ISA 7.12.2 layouts, wave32) ----
// A 16x32 f16: lane m=lane&15, half h=lane>>4; per-lane chunks K=[8h,8h+7] and K=[16+8h,16+8h+7]
__device__ __forceinline__ v16h frag_a(const _Float16* rowk, int hl) {
  union { v16h v; v8h h[2]; } u;
  u.h[0] = *(const v8h*)(rowk + 8 * hl);
  u.h[1] = *(const v8h*)(rowk + 16 + 8 * hl);
  return u.v;
}
// B 32x16 f16: lane n=lane&15, half h=lane>>4; per-lane chunks K=[8h,8h+7] and K=[16+8h,16+8h+7]
__device__ __forceinline__ v16h frag_b(const _Float16* colrow, int hl) {
  union { v16h v; v8h h[2]; } u;
  u.h[0] = *(const v8h*)(colrow + 8 * hl);
  u.h[1] = *(const v8h*)(colrow + 16 + 8 * hl);
  return u.v;
}
__device__ __forceinline__ v8f wmma16(v16h a, v16h b, v8f c) {
  return __builtin_amdgcn_wmma_f32_16x16x32_f16(false, a, false, b, (short)0, c, false, false);
}

// ---- prep: fp32->f16 weights + gathered relative-position bias [H][49][49] ----
__global__ void wattn_prep(const float* __restrict__ qkv_w, const float* __restrict__ proj_w,
                           const float* __restrict__ table,
                           _Float16* __restrict__ wqkv16, _Float16* __restrict__ wproj16,
                           float* __restrict__ bias) {
  int i = blockIdx.x * 256 + threadIdx.x;
  if (i < 3 * DIM * DIM) wqkv16[i] = (_Float16)qkv_w[i];
  if (i < DIM * DIM)     wproj16[i] = (_Float16)proj_w[i];
  if (i < NHEAD * SEQ * SEQ) {
    int h = i / (SEQ * SEQ);
    int r = i % (SEQ * SEQ);
    int a = r / SEQ, b = r % SEQ;
    int dy = a / 7 - b / 7 + 6;
    int dx = a % 7 - b % 7 + 6;
    bias[i] = table[(dy * 13 + dx) * NHEAD + h];
  }
}

// ---- fused window attention: one block per window, 8 waves, 2 heads/wave ----
__global__ void __launch_bounds__(256, 1)
wattn_main(const float* __restrict__ x, const float* __restrict__ attn_mask,
           const float* __restrict__ qkv_b, const float* __restrict__ proj_b,
           const _Float16* __restrict__ wqkv16, const _Float16* __restrict__ wproj16,
           const float* __restrict__ bias, float* __restrict__ out) {
  extern __shared__ char smem[];
  _Float16* s_x  = (_Float16*)smem;                 // [64][520] f16 window activations
  _Float16* s_ao = (_Float16*)(smem + SX_BYTES);    // [64][520] f16 attention output
  char* wavebase = smem + 2 * SX_BYTES + (threadIdx.x >> 5) * WAVE_BYTES;
  _Float16* s_q  = (_Float16*)wavebase;             // [64][32]
  _Float16* s_k  = (_Float16*)(wavebase + 4096);    // [64][32]
  _Float16* s_p  = (_Float16*)wavebase;             // [64][72]  (reuses q+k after scores)
  _Float16* s_vt = (_Float16*)(wavebase + QKP_BYTES); // [32][64] v transposed

  const int tid  = threadIdx.x;
  const int lane = tid & 31;
  const int wv   = tid >> 5;
  const int hl   = lane >> 4;   // which 16-lane half
  const int ln   = lane & 15;
  const int bwin = blockIdx.x;

  // ---- stage x (f32 -> f16, rows 49..63 zero) ----
  const float* xb = x + (size_t)bwin * SEQ * DIM;
  for (int it = tid; it < SEQP * (DIM / 4); it += 256) {
    int row = it / (DIM / 4);
    int c4  = (it % (DIM / 4)) * 4;
    float4 vx = make_float4(0.f, 0.f, 0.f, 0.f);
    if (row < SEQ) vx = *(const float4*)(xb + row * DIM + c4);
    _Float16* d = s_x + row * SXS + c4;
    d[0] = (_Float16)vx.x; d[1] = (_Float16)vx.y;
    d[2] = (_Float16)vx.z; d[3] = (_Float16)vx.w;
  }
  __syncthreads();

  const float* maskw = attn_mask + (size_t)(bwin & 63) * SEQ * SEQ;

  for (int hh = 0; hh < 2; ++hh) {
    const int h = wv + 8 * hh;
    const float* biash = bias + (size_t)h * SEQ * SEQ;

    // ---- QKV GEMM, single K-pass: 6 col-tiles (q0,q1,k0,k1,v0,v1) resident ----
    // acc[ct][mt]: ct = p*2+t (p: 0=q,1=k,2=v), 24 x v8f = 192 VGPRs
    const _Float16* wrow[6];
    float bn[6];
#pragma unroll
    for (int ct = 0; ct < 6; ++ct) {
      const int p = ct >> 1, t = ct & 1;
      const int ncol = p * DIM + h * HEADD + t * 16 + ln;
      wrow[ct] = wqkv16 + (size_t)ncol * DIM;
      bn[ct] = qkv_b[ncol];
    }
    v8f acc[6][4];
#pragma unroll
    for (int ct = 0; ct < 6; ++ct)
#pragma unroll
      for (int mt = 0; mt < 4; ++mt) acc[ct][mt] = v8f{};

#pragma unroll
    for (int kk = 0; kk < DIM; kk += 32) {
      if ((kk & 127) == 0) {
#pragma unroll
        for (int ct = 0; ct < 6; ++ct)
          __builtin_prefetch(wrow[ct] + kk + 128, 0, 3);  // global_prefetch_b8 (L2-resident stream)
      }
      v16h afr[4];
#pragma unroll
      for (int mt = 0; mt < 4; ++mt)
        afr[mt] = frag_a(s_x + (mt * 16 + ln) * SXS + kk, hl);
#pragma unroll
      for (int ct = 0; ct < 6; ++ct) {
        v16h bfr = frag_b(wrow[ct] + kk, hl);
#pragma unroll
        for (int mt = 0; mt < 4; ++mt)
          acc[ct][mt] = wmma16(afr[mt], bfr, acc[ct][mt]);
      }
    }

    // epilogue: bias (+scale for q), store q/k row-major, v transposed
#pragma unroll
    for (int ct = 0; ct < 6; ++ct) {
      const int p = ct >> 1, t = ct & 1;
      if (p < 2) {
        _Float16* dst = (p == 0) ? s_q : s_k;
        const float sc = (p == 0) ? SCALE : 1.0f;
#pragma unroll
        for (int mt = 0; mt < 4; ++mt)
#pragma unroll
          for (int r = 0; r < 8; ++r)
            dst[(mt * 16 + 8 * hl + r) * HEADD + t * 16 + ln] =
                (_Float16)((acc[ct][mt][r] + bn[ct]) * sc);
      } else {  // v: store transposed -> vT[d][m], 8 contiguous f16 per lane
#pragma unroll
        for (int mt = 0; mt < 4; ++mt) {
          v8h pk;
#pragma unroll
          for (int r = 0; r < 8; ++r) pk[r] = (_Float16)(acc[ct][mt][r] + bn[ct]);
          *(v8h*)(s_vt + (t * 16 + ln) * SEQP + mt * 16 + 8 * hl) = pk;
        }
      }
    }

    // ---- scores = q k^T (+mask+bias), softmax, P -> LDS (per 16-row band) ----
#pragma unroll
    for (int mt = 0; mt < 4; ++mt) {
      v16h afr = frag_a(s_q + (mt * 16 + ln) * HEADD, hl);
      v8f s[4];
#pragma unroll
      for (int nt = 0; nt < 4; ++nt) {
        v16h bfr = frag_b(s_k + (nt * 16 + ln) * HEADD, hl);
        s[nt] = wmma16(afr, bfr, v8f{});
      }
#pragma unroll
      for (int r = 0; r < 8; ++r) {
        const int m  = mt * 16 + 8 * hl + r;
        const int mc = (m < SEQ) ? m : SEQ - 1;
        float vals[4];
        float mx = -1e30f;
#pragma unroll
        for (int nt = 0; nt < 4; ++nt) {
          int n = nt * 16 + ln;
          float v = s[nt][r];
          v = (n < SEQ) ? (v + maskw[mc * SEQ + n] + biash[mc * SEQ + n]) : -1e30f;
          vals[nt] = v;
          mx = fmaxf(mx, v);
        }
#pragma unroll
        for (int off = 1; off < 16; off <<= 1) mx = fmaxf(mx, __shfl_xor(mx, off));
        float sum = 0.f;
#pragma unroll
        for (int nt = 0; nt < 4; ++nt) {
          float e = __expf(vals[nt] - mx);
          vals[nt] = e;
          sum += e;
        }
#pragma unroll
        for (int off = 1; off < 16; off <<= 1) sum += __shfl_xor(sum, off);
        const float inv = 1.0f / sum;
#pragma unroll
        for (int nt = 0; nt < 4; ++nt)
          s_p[m * 72 + nt * 16 + ln] = (_Float16)(vals[nt] * inv);
      }
    }

    // ---- out_h = P @ V : A = P[64][64] (K=64 -> 2 chunks), B = vT ----
#pragma unroll
    for (int mt = 0; mt < 4; ++mt) {
      v8f o[2] = {v8f{}, v8f{}};
#pragma unroll
      for (int kc = 0; kc < 2; ++kc) {
        v16h afr = frag_a(s_p + (mt * 16 + ln) * 72 + kc * 32, hl);
#pragma unroll
        for (int dt = 0; dt < 2; ++dt) {
          v16h bfr = frag_b(s_vt + (dt * 16 + ln) * SEQP + kc * 32, hl);
          o[dt] = wmma16(afr, bfr, o[dt]);
        }
      }
#pragma unroll
      for (int dt = 0; dt < 2; ++dt)
#pragma unroll
        for (int r = 0; r < 8; ++r)
          s_ao[(mt * 16 + 8 * hl + r) * SXS + h * HEADD + dt * 16 + ln] =
              (_Float16)o[dt][r];
    }
  }
  __syncthreads();

  // ---- projection GEMM, single K-pass: 4 col-tiles per wave resident ----
  float* ob = out + (size_t)bwin * SEQ * DIM;
  const _Float16* prow[4];
  float pb[4];
#pragma unroll
  for (int ntl = 0; ntl < 4; ++ntl) {
    const int n0 = (wv * 4 + ntl) * 16;
    prow[ntl] = wproj16 + (size_t)(n0 + ln) * DIM;
    pb[ntl] = proj_b[n0 + ln];
  }
  v8f acc[4][4];
#pragma unroll
  for (int ntl = 0; ntl < 4; ++ntl)
#pragma unroll
    for (int mt = 0; mt < 4; ++mt) acc[ntl][mt] = v8f{};

#pragma unroll
  for (int kk = 0; kk < DIM; kk += 32) {
    if ((kk & 127) == 0) {
#pragma unroll
      for (int ntl = 0; ntl < 4; ++ntl)
        __builtin_prefetch(prow[ntl] + kk + 128, 0, 3);
    }
    v16h afr[4];
#pragma unroll
    for (int mt = 0; mt < 4; ++mt)
      afr[mt] = frag_a(s_ao + (mt * 16 + ln) * SXS + kk, hl);
#pragma unroll
    for (int ntl = 0; ntl < 4; ++ntl) {
      v16h bfr = frag_b(prow[ntl] + kk, hl);
#pragma unroll
      for (int mt = 0; mt < 4; ++mt)
        acc[ntl][mt] = wmma16(afr[mt], bfr, acc[ntl][mt]);
    }
  }
#pragma unroll
  for (int ntl = 0; ntl < 4; ++ntl) {
    const int n0 = (wv * 4 + ntl) * 16;
#pragma unroll
    for (int mt = 0; mt < 4; ++mt)
#pragma unroll
      for (int r = 0; r < 8; ++r) {
        int m = mt * 16 + 8 * hl + r;
        if (m < SEQ) ob[m * DIM + n0 + ln] = acc[ntl][mt][r] + pb[ntl];
      }
  }
}

extern "C" void kernel_launch(void* const* d_in, const int* in_sizes, int n_in,
                              void* d_out, int out_size, void* d_ws, size_t ws_size,
                              hipStream_t stream) {
  const float* x      = (const float*)d_in[0];
  const float* mask   = (const float*)d_in[1];
  const float* qkv_w  = (const float*)d_in[2];
  const float* qkv_b  = (const float*)d_in[3];
  const float* proj_w = (const float*)d_in[4];
  const float* proj_b = (const float*)d_in[5];
  const float* table  = (const float*)d_in[6];
  float* out = (float*)d_out;

  char* ws = (char*)d_ws;
  _Float16* wqkv16  = (_Float16*)ws;                            // 1,572,864 B
  _Float16* wproj16 = (_Float16*)(ws + 3 * DIM * DIM * 2);      //   524,288 B
  float*    bias    = (float*)(ws + 3 * DIM * DIM * 2 + DIM * DIM * 2);  // 153,664 B

  (void)in_sizes; (void)n_in; (void)out_size; (void)ws_size;

  hipFuncSetAttribute((const void*)wattn_main,
                      hipFuncAttributeMaxDynamicSharedMemorySize, SMEM_TOTAL);

  wattn_prep<<<(3 * DIM * DIM + 255) / 256, 256, 0, stream>>>(
      qkv_w, proj_w, table, wqkv16, wproj16, bias);
  wattn_main<<<4096, 256, SMEM_TOTAL, stream>>>(
      x, mask, qkv_b, proj_b, wqkv16, wproj16, bias, out);
}
